// CustomGNN_54503134986745
// MI455X (gfx1250) — matrix operand order repaired and checked by reading, not compiled, not run on hardware
//
#include <hip/hip_runtime.h>
#include <hip/hip_bf16.h>

#define NN   50000
#define EE   800000
#define DDIM 128
#define LLAY 3
#define GG   64
#define DOUT 10

typedef __attribute__((ext_vector_type(16))) _Float16 v16h;
typedef __attribute__((ext_vector_type(8)))  _Float16 v8h;
typedef __attribute__((ext_vector_type(8)))  float    v8f;

// ---------------------------------------------------------------- utilities
__global__ void zero_kernel(float* __restrict__ p, size_t n) {
  size_t i = (size_t)blockIdx.x * blockDim.x + threadIdx.x;
  size_t stride = (size_t)gridDim.x * blockDim.x;
  for (; i < n; i += stride) p[i] = 0.f;
}

// Repack lin_w [L][4][128][128] (f32, d-major rows, o contiguous) into per-lane
// WMMA B fragments (f16). Fragment id = ((l*32 + j)*4 + c), 32 lanes x 16 halves.
// Lane layout (contiguous k order, same as A in gemm): lane<16 -> N=lane, K=c*32+h ; lane>=16 -> N=lane-16, K=c*32+16+h.
__global__ void repack_w_kernel(const float* __restrict__ lin_w, _Float16* __restrict__ Wp) {
  int idx = blockIdx.x * 256 + threadIdx.x;           // 0 .. L*32*4*32-1
  if (idx >= LLAY * 32 * 4 * 32) return;
  int lane = idx & 31;
  int c    = (idx >> 5) & 3;
  int j    = (idx >> 7) & 31;
  int l    = idx >> 12;
  int kk   = j >> 3;                                   // which of A,B,D,E
  int o    = ((j & 7) << 4) + (lane & 15);             // output column in matrix
  int kbase = c * 32 + ((lane >> 4) << 4);
  const float* W = lin_w + ((size_t)(l * 4 + kk)) * DDIM * DDIM;
  _Float16* dst = Wp + (size_t)idx * 16;
#pragma unroll
  for (int h = 0; h < 16; ++h)
    dst[h] = (_Float16)W[(size_t)(kbase + h) * DDIM + o];
}

// ------------------------------------------------------- fused 4-way projection
// proj[k][n][o] = sum_d x[n][d] * W_k[d][o] + b_k[o]   via v_wmma_f32_16x16x32_f16
// Block: 16 rows of x, 8 waves; wave w covers columns [64w, 64w+64) of 512.
__global__ __launch_bounds__(256) void gemm_proj_kernel(
    const float* __restrict__ X, const _Float16* __restrict__ WpL,
    const float* __restrict__ Lb, float* __restrict__ PROJ) {
  __shared__ __align__(16) _Float16 sA[16 * DDIM];

  const int tid = threadIdx.x;
  const float* xt = X + (size_t)blockIdx.x * 16 * DDIM;
#pragma unroll
  for (int i = 0; i < 2; ++i) {                        // 512 float4 loads total
    int f = tid + i * 256;
    float4 v = ((const float4*)xt)[f];
    int b = f * 4;
    sA[b + 0] = (_Float16)v.x; sA[b + 1] = (_Float16)v.y;
    sA[b + 2] = (_Float16)v.z; sA[b + 3] = (_Float16)v.w;
  }
  __syncthreads();

  const int wave = tid >> 5;
  const int lane = tid & 31;
  const int mrow = lane & 15;
  const int alo  = (lane < 16) ? 0 : 16;               // contiguous k: lanes 16-31 start at k+16

  v16h a[4];
#pragma unroll
  for (int c = 0; c < 4; ++c) {
    v8h p0 = *(const v8h*)&sA[mrow * DDIM + c * 32 + alo];
    v8h p1 = *(const v8h*)&sA[mrow * DDIM + c * 32 + 8 + alo];
#pragma unroll
    for (int h = 0; h < 8; ++h) { a[c][h] = p0[h]; a[c][8 + h] = p1[h]; }
  }

#pragma unroll
  for (int t = 0; t < 4; ++t) {
    const int j = wave * 4 + t;                        // N-tile 0..31 of 512 cols
    v8f acc = {};
#pragma unroll
    for (int c = 0; c < 4; ++c) {
      const _Float16* bp = WpL + ((size_t)(j * 4 + c) * 32 + lane) * 16;
      v8h b0 = *(const v8h*)bp;
      v8h b1 = *(const v8h*)(bp + 8);
      v16h bf;
#pragma unroll
      for (int h = 0; h < 8; ++h) { bf[h] = b0[h]; bf[8 + h] = b1[h]; }
      acc = __builtin_amdgcn_wmma_f32_16x16x32_f16(false, a[c], false, bf,
                                                   (short)0, acc, false, false);
    }
    const int kk   = j >> 3;
    const int ocol = ((j & 7) << 4) + (lane & 15);
    const float bias = Lb[kk * DDIM + ocol];
    float* outp = PROJ + ((size_t)kk * NN + (size_t)blockIdx.x * 16) * DDIM + ocol;
    const int mhi = (lane >> 4) << 3;                  // C layout: lanes 16-31 -> M += 8
#pragma unroll
    for (int r = 0; r < 8; ++r)
      outp[(size_t)(r + mhi) * DDIM] = acc[r] + bias;
  }
}

// ------------------------------------------------------------- edge scatter
// one wave per edge; lane handles 4 channels
__global__ __launch_bounds__(256) void edge_kernel(
    const int* __restrict__ ei, const float* __restrict__ Bx,
    const float* __restrict__ Dx, const float* __restrict__ Ex,
    float* __restrict__ num, float* __restrict__ den) {
  const int e    = blockIdx.x * 8 + (threadIdx.x >> 5);
  const int lane = threadIdx.x & 31;
  const int js = ei[e];        // src
  const int id = ei[EE + e];   // dst
  const int ch = lane * 4;
  float4 d4 = *(const float4*)(Dx + (size_t)id * DDIM + ch);
  float4 e4 = *(const float4*)(Ex + (size_t)js * DDIM + ch);
  float4 b4 = *(const float4*)(Bx + (size_t)js * DDIM + ch);
  float su[4] = {d4.x + e4.x, d4.y + e4.y, d4.z + e4.z, d4.w + e4.w};
  float bb[4] = {b4.x, b4.y, b4.z, b4.w};
  float* np = num + (size_t)id * DDIM + ch;
  float* dp = den + (size_t)id * DDIM + ch;
#pragma unroll
  for (int t = 0; t < 4; ++t) {
    float sg = 1.f / (1.f + __expf(-su[t]));
    atomicAdd(np + t, sg * bb[t]);
    atomicAdd(dp + t, sg);
  }
}

// --------------------------------------------- batchnorm stats (sum, sumsq)
__global__ __launch_bounds__(256) void stats_kernel(
    const float* __restrict__ Ax, const float* __restrict__ num,
    const float* __restrict__ den, float* __restrict__ bnstats) {
  __shared__ float sSum[DDIM], sSq[DDIM];
  if (threadIdx.x < DDIM) { sSum[threadIdx.x] = 0.f; sSq[threadIdx.x] = 0.f; }
  __syncthreads();
  const size_t total = (size_t)NN * DDIM / 4;
  for (size_t i = (size_t)blockIdx.x * 256 + threadIdx.x; i < total;
       i += (size_t)gridDim.x * 256) {
    float4 a  = ((const float4*)Ax)[i];
    float4 nm = ((const float4*)num)[i];
    float4 dn = ((const float4*)den)[i];
    int ch = (int)((i * 4) & (DDIM - 1));
    float h[4] = {a.x + nm.x / (dn.x + 1e-6f), a.y + nm.y / (dn.y + 1e-6f),
                  a.z + nm.z / (dn.z + 1e-6f), a.w + nm.w / (dn.w + 1e-6f)};
#pragma unroll
    for (int t = 0; t < 4; ++t) {
      atomicAdd(&sSum[ch + t], h[t]);
      atomicAdd(&sSq[ch + t], h[t] * h[t]);
    }
  }
  __syncthreads();
  if (threadIdx.x < DDIM) {
    atomicAdd(&bnstats[threadIdx.x], sSum[threadIdx.x]);
    atomicAdd(&bnstats[DDIM + threadIdx.x], sSq[threadIdx.x]);
  }
}

// ------------------------------------- BN apply + ReLU + residual, in place
__global__ __launch_bounds__(256) void update_kernel(
    float* __restrict__ X, const float* __restrict__ Ax,
    const float* __restrict__ num, const float* __restrict__ den,
    const float* __restrict__ bnstats, const float* __restrict__ gamma,
    const float* __restrict__ beta) {
  const size_t i = (size_t)blockIdx.x * 256 + threadIdx.x;   // float4 index (exact)
  float4 a  = ((const float4*)Ax)[i];
  float4 nm = ((const float4*)num)[i];
  float4 dn = ((const float4*)den)[i];
  float4 xv = ((const float4*)X)[i];
  int ch = (int)((i * 4) & (DDIM - 1));
  float hv[4]  = {a.x + nm.x / (dn.x + 1e-6f), a.y + nm.y / (dn.y + 1e-6f),
                  a.z + nm.z / (dn.z + 1e-6f), a.w + nm.w / (dn.w + 1e-6f)};
  float xin[4] = {xv.x, xv.y, xv.z, xv.w};
  float out[4];
#pragma unroll
  for (int t = 0; t < 4; ++t) {
    float mu  = bnstats[ch + t] * (1.f / NN);
    float var = bnstats[DDIM + ch + t] * (1.f / NN) - mu * mu;
    float nrm = (hv[t] - mu) * rsqrtf(var + 1e-5f) * gamma[ch + t] + beta[ch + t];
    out[t] = xin[t] + fmaxf(nrm, 0.f);
  }
  ((float4*)X)[i] = make_float4(out[0], out[1], out[2], out[3]);
}

// ----------------------------------------------------------------- pooling
__global__ void count_kernel(const int* __restrict__ batch, float* __restrict__ counts) {
  int n = blockIdx.x * 256 + threadIdx.x;
  if (n < NN) atomicAdd(&counts[batch[n]], 1.f);
}

__global__ __launch_bounds__(256) void pool_kernel(
    const float* __restrict__ X, const int* __restrict__ batch,
    float* __restrict__ pooled) {
  size_t i = (size_t)blockIdx.x * 256 + threadIdx.x;   // float4 index (exact)
  int n  = (int)(i >> 5);
  int ch = (int)((i * 4) & (DDIM - 1));
  int g  = batch[n];
  float4 v = ((const float4*)X)[i];
  float* p = pooled + (size_t)g * DDIM + ch;
  atomicAdd(p + 0, v.x); atomicAdd(p + 1, v.y);
  atomicAdd(p + 2, v.z); atomicAdd(p + 3, v.w);
}

__global__ void pool_div_kernel(float* __restrict__ pooled, const float* __restrict__ counts) {
  int i = blockIdx.x * 256 + threadIdx.x;              // 8192
  int g = i >> 7;
  pooled[i] /= fmaxf(counts[g], 1.f);
}

// --------------------------------------------------------------- final MLP
__global__ __launch_bounds__(256) void mlp_kernel(
    const float* __restrict__ pooled, const float* __restrict__ W1,
    const float* __restrict__ b1, const float* __restrict__ W2,
    const float* __restrict__ b2, float* __restrict__ out) {
  __shared__ float sH[GG * DDIM];
  for (int idx = threadIdx.x; idx < GG * DDIM; idx += 256) {
    int g = idx >> 7, o = idx & 127;
    float acc = b1[o];
    for (int d = 0; d < DDIM; ++d) acc += pooled[g * DDIM + d] * W1[d * DDIM + o];
    sH[idx] = fmaxf(acc, 0.f);
  }
  __syncthreads();
  for (int idx = threadIdx.x; idx < GG * DOUT; idx += 256) {
    int g = idx / DOUT, q = idx % DOUT;
    float acc = b2[q];
    for (int o = 0; o < DDIM; ++o) acc += sH[g * DDIM + o] * W2[o * DOUT + q];
    out[idx] = acc;
  }
}

// ----------------------------------------------------------------- launcher
extern "C" void kernel_launch(void* const* d_in, const int* in_sizes, int n_in,
                              void* d_out, int out_size, void* d_ws, size_t ws_size,
                              hipStream_t stream) {
  const float* x     = (const float*)d_in[0];
  // d_in[1] = edge_attr: never read in forward (edge features overwritten, unused)
  const int*   ei    = (const int*)d_in[2];
  const int*   batch = (const int*)d_in[3];
  const float* lin_w = (const float*)d_in[4];
  const float* lin_b = (const float*)d_in[5];
  const float* gamma = (const float*)d_in[6];
  const float* beta  = (const float*)d_in[7];
  const float* w1    = (const float*)d_in[8];
  const float* b1    = (const float*)d_in[9];
  const float* w2    = (const float*)d_in[10];
  const float* b2    = (const float*)d_in[11];

  char* base = (char*)d_ws; size_t off = 0;
  auto take = [&](size_t bytes) -> char* {
    char* p = base + off; off += (bytes + 255) & ~(size_t)255; return p;
  };
  float*    Xb     = (float*)take((size_t)NN * DDIM * 4);
  float*    PROJ   = (float*)take((size_t)4 * NN * DDIM * 4);
  float*    NUMb   = (float*)take((size_t)NN * DDIM * 4);   // NUM, DEN, STATS contiguous
  float*    DENb   = (float*)take((size_t)NN * DDIM * 4);
  float*    STATS  = (float*)take(2 * DDIM * 4);
  _Float16* Wp     = (_Float16*)take((size_t)LLAY * 32 * 4 * 32 * 16 * 2);
  float*    POOLED = (float*)take((size_t)GG * DDIM * 4);   // POOLED, COUNTS contiguous
  float*    COUNTS = (float*)take((size_t)GG * 4);

  hipMemcpyAsync(Xb, x, (size_t)NN * DDIM * 4, hipMemcpyDeviceToDevice, stream);
  repack_w_kernel<<<48, 256, 0, stream>>>(lin_w, Wp);

  for (int l = 0; l < LLAY; ++l) {
    zero_kernel<<<2048, 256, 0, stream>>>(NUMb, (size_t)2 * NN * DDIM + 2 * DDIM);
    gemm_proj_kernel<<<NN / 16, 256, 0, stream>>>(
        Xb, Wp + (size_t)l * 32 * 4 * 32 * 16, lin_b + l * 4 * DDIM, PROJ);
    edge_kernel<<<EE / 8, 256, 0, stream>>>(
        ei, PROJ + (size_t)1 * NN * DDIM, PROJ + (size_t)2 * NN * DDIM,
        PROJ + (size_t)3 * NN * DDIM, NUMb, DENb);
    stats_kernel<<<1024, 256, 0, stream>>>(PROJ, NUMb, DENb, STATS);
    update_kernel<<<NN * DDIM / 4 / 256, 256, 0, stream>>>(
        Xb, PROJ, NUMb, DENb, STATS, gamma + l * DDIM, beta + l * DDIM);
  }

  zero_kernel<<<64, 256, 0, stream>>>(POOLED, (size_t)GG * DDIM + GG);
  count_kernel<<<(NN + 255) / 256, 256, 0, stream>>>(batch, COUNTS);
  pool_kernel<<<NN * DDIM / 4 / 256, 256, 0, stream>>>(Xb, batch, POOLED);
  pool_div_kernel<<<GG * DDIM / 256, 256, 0, stream>>>(POOLED, COUNTS);
  mlp_kernel<<<1, 256, 0, stream>>>(POOLED, w1, b1, w2, b2, (float*)d_out);

  (void)in_sizes; (void)n_in; (void)out_size; (void)ws_size;
}
